// LoopMoE_84851373900524
// MI455X (gfx1250) — matrix operand order repaired and checked by reading, not compiled, not run on hardware
//
#include <hip/hip_runtime.h>
#include <hip/hip_bf16.h>

// ---------------------------------------------------------------------------
// MoE (top-2 of 8 experts, SwiGLU) for MI455X / gfx1250.
// Router builds per-expert token gather lists; each block runs 32 gathered
// rows of one expert with V_WMMA_F32_16X16X32_BF16 (fp32 accumulation).
// Two M-tiles per wave share every B fragment -> 2x less operand traffic.
// ---------------------------------------------------------------------------

typedef __attribute__((ext_vector_type(16))) __bf16 v16bf;
typedef __attribute__((ext_vector_type(8)))  float  v8f;

#define HH 1024   // hidden dim
#define II 2048   // intermediate dim
#define EE 8      // experts
#define TM 32     // gathered rows per block (2 WMMA M tiles)

static __device__ __forceinline__ __bf16 f2bf(float f) {
  unsigned u = __builtin_bit_cast(unsigned, f);
  u += 0x7FFFu + ((u >> 16) & 1u);          // round-to-nearest-even
  unsigned short h = (unsigned short)(u >> 16);
  return __builtin_bit_cast(__bf16, h);
}

static __device__ __forceinline__ unsigned pack2(float a, float b) {
  unsigned ua = __builtin_bit_cast(unsigned, a);
  unsigned ub = __builtin_bit_cast(unsigned, b);
  ua += 0x7FFFu + ((ua >> 16) & 1u);
  ub += 0x7FFFu + ((ub >> 16) & 1u);
  return (ua >> 16) | (ub & 0xFFFF0000u);
}

struct U32x8 { uint4 a, b; };

// A-fragment (16x32 bf16): lane row M=lane&15, runs K=k0..k0+7, k0+16..k0+23.
static __device__ __forceinline__ v16bf fragA_lds(const __bf16* p0, const __bf16* p1) {
  U32x8 u;
  u.a = *(const uint4*)p0;
  u.b = *(const uint4*)p1;
  return __builtin_bit_cast(v16bf, u);
}

// B-fragment (32x16 bf16): lane col N=lane&15, runs K=k0..k0+7, k0+16..k0+23.
static __device__ __forceinline__ v16bf fragB_bf(const __bf16* p) {
  U32x8 u;
  u.a = *(const uint4*)p;
  u.b = *(const uint4*)(p + 16);
  return __builtin_bit_cast(v16bf, u);
}

static __device__ __forceinline__ v16bf fragB_f32(const float* p) {
  const float4* q = (const float4*)p;
  float4 f0 = q[0], f1 = q[1], f2 = q[4], f3 = q[5];
  v16bf r;
  r[0]  = f2bf(f0.x); r[1]  = f2bf(f0.y); r[2]  = f2bf(f0.z); r[3]  = f2bf(f0.w);
  r[4]  = f2bf(f1.x); r[5]  = f2bf(f1.y); r[6]  = f2bf(f1.z); r[7]  = f2bf(f1.w);
  r[8]  = f2bf(f2.x); r[9]  = f2bf(f2.y); r[10] = f2bf(f2.z); r[11] = f2bf(f2.w);
  r[12] = f2bf(f3.x); r[13] = f2bf(f3.y); r[14] = f2bf(f3.z); r[15] = f2bf(f3.w);
  return r;
}

// ---------------------------------------------------------------------------
// fp32 -> bf16 bulk converter: thread i converts 8 floats -> one uint4.
// ---------------------------------------------------------------------------
__global__ __launch_bounds__(256) void cvt_bf16_kernel(
    const float* __restrict__ src, uint4* __restrict__ dst, long n8) {
  long i = (long)blockIdx.x * blockDim.x + threadIdx.x;
  if (i >= n8) return;
  const float4* q = (const float4*)src + i * 2;
  float4 a = q[0], b = q[1];
  uint4 o;
  o.x = pack2(a.x, a.y); o.y = pack2(a.z, a.w);
  o.z = pack2(b.x, b.y); o.w = pack2(b.z, b.w);
  dst[i] = o;
}

// ---------------------------------------------------------------------------
// Router: one wave32 per token; softmax + top-2; appends (token, prob) to the
// two selected experts' gather lists via atomics.
// ---------------------------------------------------------------------------
__global__ __launch_bounds__(256) void router_kernel(
    const float* __restrict__ x, const float* __restrict__ rw,
    int* __restrict__ cnt, int* __restrict__ tlist, float* __restrict__ wlist,
    int T) {
  const int lane = threadIdx.x & 31;
  const int wv   = blockIdx.x * (blockDim.x >> 5) + (threadIdx.x >> 5);
  if (wv >= T) return;
  const float* xt = x + (size_t)wv * HH;

  float acc[EE];
#pragma unroll
  for (int e = 0; e < EE; ++e) acc[e] = 0.f;
  for (int j = 0; j < HH; j += 32) {
    float xv = xt[j + lane];
#pragma unroll
    for (int e = 0; e < EE; ++e) acc[e] += xv * rw[e * HH + j + lane];
  }
#pragma unroll
  for (int e = 0; e < EE; ++e) {
#pragma unroll
    for (int off = 16; off > 0; off >>= 1) acc[e] += __shfl_xor(acc[e], off, 32);
  }
  if (lane == 0) {
    float mx = acc[0];
#pragma unroll
    for (int e = 1; e < EE; ++e) mx = fmaxf(mx, acc[e]);
    float p[EE], s = 0.f;
#pragma unroll
    for (int e = 0; e < EE; ++e) { p[e] = __expf(acc[e] - mx); s += p[e]; }
    float inv = __builtin_amdgcn_rcpf(s);
#pragma unroll
    for (int e = 0; e < EE; ++e) p[e] *= inv;
    int i1 = 0;
#pragma unroll
    for (int e = 1; e < EE; ++e) if (p[e] > p[i1]) i1 = e;
    int i2 = (i1 == 0) ? 1 : 0;
#pragma unroll
    for (int e = 0; e < EE; ++e) if (e != i1 && p[e] > p[i2]) i2 = e;
    int pos1 = atomicAdd(&cnt[i1], 1);
    tlist[(size_t)i1 * T + pos1] = wv;
    wlist[(size_t)i1 * T + pos1] = p[i1];
    int pos2 = atomicAdd(&cnt[i2], 1);
    tlist[(size_t)i2 * T + pos2] = wv;
    wlist[(size_t)i2 * T + pos2] = p[i2];
  }
}

// ---------------------------------------------------------------------------
// Fused SwiGLU expert over gathered rows.  Grid = (T/TM slots, E experts),
// slots fastest so one expert's weights stay L2-resident.  256 thr = 8 waves;
// each wave owns a 16-wide N slice and BOTH M tiles (B fragments reused 2x).
// ---------------------------------------------------------------------------
template <bool PRE>
__global__ __launch_bounds__(256) void moe_kernel(
    const float*  __restrict__ xf,  const __bf16* __restrict__ xb,
    const float*  __restrict__ w1f, const __bf16* __restrict__ w1b,
    const float*  __restrict__ w2f, const __bf16* __restrict__ w2b,
    const int*    __restrict__ cnt, const int* __restrict__ tlist,
    const float*  __restrict__ wlist, float* __restrict__ out, int T) {
  const int e     = blockIdx.y;
  const int base  = blockIdx.x * TM;
  const int count = cnt[e];
  if (base >= count) return;             // no rows for this slot

  const int tid  = threadIdx.x;
  const int wave = tid >> 5;
  const int lane = tid & 31;
  const int n    = lane & 15;            // N column / A row within tile
  const int hi   = lane >> 4;
  const int k0A  = hi * 8;               // A-fragment K base
  const int k0B  = hi * 8;               // B-fragment K base
  const int mrow = hi * 8;               // C/D: M = vgpr + mrow

  __shared__ int    toks[TM];
  __shared__ float  ews[TM];
  __shared__ __align__(16) __bf16 xs[TM][HH + 8];   // gathered x,  bf16 (66 KB)
  __shared__ __align__(16) __bf16 av[TM][II + 8];   // activation,  bf16 (132 KB)

  if (tid < TM) {
    int idx = base + tid;
    bool ok = idx < count;
    toks[tid] = tlist[(size_t)e * T + (ok ? idx : base)];   // clamp (weight 0)
    ews[tid]  = ok ? wlist[(size_t)e * T + idx] : 0.f;
  }
  __syncthreads();

  // Gather x rows -> LDS bf16 (each row is a contiguous 2KB stream).
  if (PRE) {
    for (int i = tid; i < TM * (HH / 8); i += 256) {
      int r = i >> 7;                    // 128 uint4 per row
      int c = (i & 127) * 8;
      *(uint4*)&xs[r][c] =
          ((const uint4*)(xb + (size_t)toks[r] * HH))[c / 8];
    }
  } else {
    for (int i = tid; i < TM * HH; i += 256) {
      int r = i >> 10;
      int c = i & (HH - 1);
      xs[r][c] = f2bf(xf[(size_t)toks[r] * HH + c]);
    }
  }
  __syncthreads();

  // ---- Phase 1: act = silu(x @ W1g^T) * (x @ W1l^T), [TM, II] in LDS ----
  const size_t w1off = (size_t)e * 2 * II * HH;
  for (int nt = wave; nt < II / 16; nt += 8) {
    const int nb = nt * 16;
    const size_t rg = w1off + (size_t)(nb + n) * HH;
    const size_t rl = w1off + (size_t)(II + nb + n) * HH;
    const __bf16* wgb = w1b + rg; const float* wgf = w1f + rg;
    const __bf16* wlb = w1b + rl; const float* wlf = w1f + rl;
    if (PRE) { __builtin_prefetch(wgb, 0, 0); __builtin_prefetch(wlb, 0, 0); }
    v8f ag0 = {}, ag1 = {}, al0 = {}, al1 = {};
    for (int kb = 0; kb < HH; kb += 32) {
      v16bf a0 = fragA_lds(&xs[n][kb + k0A],      &xs[n][kb + k0A + 16]);
      v16bf a1 = fragA_lds(&xs[16 + n][kb + k0A], &xs[16 + n][kb + k0A + 16]);
      v16bf bg, bl;
      if (PRE) { bg = fragB_bf(wgb + kb + k0B);  bl = fragB_bf(wlb + kb + k0B); }
      else     { bg = fragB_f32(wgf + kb + k0B); bl = fragB_f32(wlf + kb + k0B); }
      ag0 = __builtin_amdgcn_wmma_f32_16x16x32_bf16(false, a0, false, bg, (short)0, ag0, false, false);
      ag1 = __builtin_amdgcn_wmma_f32_16x16x32_bf16(false, a1, false, bg, (short)0, ag1, false, false);
      al0 = __builtin_amdgcn_wmma_f32_16x16x32_bf16(false, a0, false, bl, (short)0, al0, false, false);
      al1 = __builtin_amdgcn_wmma_f32_16x16x32_bf16(false, a1, false, bl, (short)0, al1, false, false);
    }
#pragma unroll
    for (int r = 0; r < 8; ++r) {
      float g0 = ag0[r], g1 = ag1[r];
      float s0 = __builtin_amdgcn_rcpf(1.f + __expf(-g0));   // v_rcp_f32
      float s1 = __builtin_amdgcn_rcpf(1.f + __expf(-g1));
      av[r + mrow][nb + n]      = f2bf(g0 * s0 * al0[r]);    // silu(g)*l
      av[16 + r + mrow][nb + n] = f2bf(g1 * s1 * al1[r]);
    }
  }
  __syncthreads();

  // ---- Phase 2: y = act @ W2^T, scale by gate prob, scatter-add ----
  const size_t w2off = (size_t)e * HH * II;
  for (int nt = wave; nt < HH / 16; nt += 8) {
    const int nb = nt * 16;
    const size_t r2 = w2off + (size_t)(nb + n) * II;
    const __bf16* w2rb = w2b + r2; const float* w2rf = w2f + r2;
    if (PRE) __builtin_prefetch(w2rb, 0, 0);
    v8f ac0 = {}, ac1 = {};
    for (int kb = 0; kb < II; kb += 32) {
      v16bf a0 = fragA_lds(&av[n][kb + k0A],      &av[n][kb + k0A + 16]);
      v16bf a1 = fragA_lds(&av[16 + n][kb + k0A], &av[16 + n][kb + k0A + 16]);
      v16bf b;
      if (PRE) b = fragB_bf(w2rb + kb + k0B);
      else     b = fragB_f32(w2rf + kb + k0B);
      ac0 = __builtin_amdgcn_wmma_f32_16x16x32_bf16(false, a0, false, b, (short)0, ac0, false, false);
      ac1 = __builtin_amdgcn_wmma_f32_16x16x32_bf16(false, a1, false, b, (short)0, ac1, false, false);
    }
#pragma unroll
    for (int r = 0; r < 8; ++r) {
      int m0 = r + mrow, m1 = 16 + r + mrow;
      unsafeAtomicAdd(&out[(size_t)toks[m0] * HH + nb + n], ac0[r] * ews[m0]);
      unsafeAtomicAdd(&out[(size_t)toks[m1] * HH + nb + n], ac1[r] * ews[m1]);
    }
  }
}

// ---------------------------------------------------------------------------
extern "C" void kernel_launch(void* const* d_in, const int* in_sizes, int n_in,
                              void* d_out, int out_size, void* d_ws, size_t ws_size,
                              hipStream_t stream) {
  const float* x  = (const float*)d_in[0];   // [T, H]
  const float* w1 = (const float*)d_in[1];   // [E, 2I, H]
  const float* w2 = (const float*)d_in[2];   // [E, H, I]
  const float* rw = (const float*)d_in[3];   // [E, H]
  float* out = (float*)d_out;

  const int T = in_sizes[0] / HH;

  // Workspace: [cnt 8i pad256][tlist E*T i32][wlist E*T f32][xb][w1b][w2b]
  const size_t cntB = 256;
  const size_t tlB  = (size_t)EE * T * sizeof(int);
  const size_t wlB  = (size_t)EE * T * sizeof(float);
  const size_t xbN  = (size_t)T * HH;
  const size_t w1N  = (size_t)EE * 2 * II * HH;
  const size_t w2N  = (size_t)EE * HH * II;
  const size_t headB = cntB + tlB + wlB;
  const size_t need  = headB + (xbN + w1N + w2N) * sizeof(__bf16);

  char*  wsp   = (char*)d_ws;
  int*   cnt   = (int*)wsp;
  int*   tlist = (int*)(wsp + cntB);
  float* wlist = (float*)(wsp + cntB + tlB);

  hipMemsetAsync(d_out, 0, (size_t)out_size * sizeof(float), stream);
  hipMemsetAsync(wsp, 0, cntB, stream);

  router_kernel<<<dim3((T + 7) / 8), 256, 0, stream>>>(x, rw, cnt, tlist, wlist, T);

  if (ws_size >= need) {
    __bf16* xb  = (__bf16*)(wsp + headB);
    __bf16* w1b = xb + xbN;
    __bf16* w2b = w1b + w1N;
    cvt_bf16_kernel<<<dim3((unsigned)((xbN / 8 + 255) / 256)), 256, 0, stream>>>(
        x, (uint4*)xb, (long)(xbN / 8));
    cvt_bf16_kernel<<<dim3((unsigned)((w1N / 8 + 255) / 256)), 256, 0, stream>>>(
        w1, (uint4*)w1b, (long)(w1N / 8));
    cvt_bf16_kernel<<<dim3((unsigned)((w2N / 8 + 255) / 256)), 256, 0, stream>>>(
        w2, (uint4*)w2b, (long)(w2N / 8));
    moe_kernel<true><<<dim3((T + TM - 1) / TM, EE), 256, 0, stream>>>(
        x, xb, w1, w1b, w2, w2b, cnt, tlist, wlist, out, T);
  } else {
    moe_kernel<false><<<dim3((T + TM - 1) / TM, EE), 256, 0, stream>>>(
        x, nullptr, w1, nullptr, w2, nullptr, cnt, tlist, wlist, out, T);
  }
}
